// AdjMatmulLayer_85615878078605
// MI455X (gfx1250) — matrix operand order repaired and checked by reading, not compiled, not run on hardware
//
#include <hip/hip_runtime.h>
#include <math.h>

// ---------------------------------------------------------------------------
// Problem constants (reference: HIDDEN=512, HEADS=8, n=128, bsz=4)
// ---------------------------------------------------------------------------
#define NSEQ   128
#define BSZ    4
#define HID    512
#define NH     8
#define DH     64
#define MTOK   (NSEQ * NSEQ * BSZ)      // 65536 token rows (i,t,b)
#define LN_EPS 1e-5f

// ---------------------------------------------------------------------------
// CDNA5 WMMA plumbing (wave32, V_WMMA_F32_16X16X32_BF16)
// ---------------------------------------------------------------------------
typedef __attribute__((ext_vector_type(16))) __bf16 v16bf;
typedef __attribute__((ext_vector_type(8)))  float  v8f;
typedef __attribute__((ext_vector_type(4)))  unsigned int u32x4;   // POD 16B chunk
typedef int gv4i __attribute__((vector_size(16)));                 // builtin's pointee

union BFU {
    u32x4          u[2];
    unsigned short s[16];
    v16bf          v;
};

static __device__ __forceinline__ v8f wmma_bf16(v16bf a, v16bf b, v8f c) {
    return __builtin_amdgcn_wmma_f32_16x16x32_bf16(
        /*neg_a=*/false, a, /*neg_b=*/false, b,
        /*c_mod=*/(short)0, c, /*reuse_a=*/false, /*reuse_b=*/false);
}

// fp32 -> bf16 round-to-nearest-even, stored as ushort
static __device__ __forceinline__ unsigned short f2bf(float f) {
    unsigned u = __float_as_uint(f);
    return (unsigned short)((u + 0x7fffu + ((u >> 16) & 1u)) >> 16);
}

// A fragment: 16x32 bf16, row-major source, rows = row0+lane%16.
// Layout: lanes 0-15 hold K = k0+[0..15];
//         lanes 16-31 hold K = k0+[16..31] (matches B fragments).
static __device__ __forceinline__ v16bf loadA(const unsigned short* base,
                                              long lda, int row0, int k0,
                                              int half, int lr) {
    const unsigned short* p = base + (size_t)(row0 + lr) * lda + k0 + half * 16;
    BFU u;
    u.u[0] = *reinterpret_cast<const u32x4*>(p);
    u.u[1] = *reinterpret_cast<const u32x4*>(p + 8);
    return u.v;
}

// B fragment from a "weight-form" matrix W[N][K] (row n contiguous over K):
// computes C = A @ W^T. Lane holds column n = n0+lane%16,
// K = k0 + half*16 + [0..15] contiguous.
static __device__ __forceinline__ v16bf loadBW(const unsigned short* base,
                                               long ldb, int n0, int k0,
                                               int half, int lr) {
    const unsigned short* p = base + (size_t)(n0 + lr) * ldb + k0 + half * 16;
    BFU u;
    u.u[0] = *reinterpret_cast<const u32x4*>(p);
    u.u[1] = *reinterpret_cast<const u32x4*>(p + 8);
    return u.v;
}

// B fragment from an LDS tile Bs[32][HID] holding B in K-major (k = row, n = col):
// lane holds column n, K = half*16 + m  (strided ds_load_u16 reads).
static __device__ __forceinline__ v16bf loadB_lds(const unsigned short* Bs,
                                                  int n, int half) {
    BFU u;
#pragma unroll
    for (int m = 0; m < 16; ++m)
        u.s[m] = Bs[(size_t)(half * 16 + m) * HID + n];
    return u.v;
}

#ifndef __has_builtin
#define __has_builtin(x) 0
#endif
#if __has_builtin(__builtin_amdgcn_global_load_async_to_lds_b128) && \
    __has_builtin(__builtin_amdgcn_s_wait_asynccnt)
#define HAVE_ASYNC_LDS 1
#else
#define HAVE_ASYNC_LDS 0
#endif

// ---------------------------------------------------------------------------
// K0: fp32 -> bf16 bulk convert (4 elems / thread)
// ---------------------------------------------------------------------------
__global__ void cvt_f32_bf16(const float* __restrict__ src,
                             unsigned short* __restrict__ dst, int n4) {
    int gid = blockIdx.x * blockDim.x + threadIdx.x;
    if (gid < n4) {
        int i = gid * 4;
#pragma unroll
        for (int e = 0; e < 4; ++e) dst[i + e] = f2bf(src[i + e]);
    }
}

// ---------------------------------------------------------------------------
// K1: fused q/k projection.  A = state_bf16 (MTOK x 512, row-major),
// W-form B = Wq or Wk (512 x 512).  Each wave register-blocks 4 N-tiles
// (64 columns), reusing one A fragment across 4 WMMAs per k-step.
// grid = (MTOK/16, 2); blockIdx.y: 0 -> q, 1 -> k.  block = 256 (8 waves).
// Scatters results into q2/k2[(i*B+b)*NH+hd][t][d] bf16 layout.
// ---------------------------------------------------------------------------
__global__ void qk_kernel(const unsigned short* __restrict__ sbf,
                          const unsigned short* __restrict__ wqb,
                          const unsigned short* __restrict__ wkb,
                          const float* __restrict__ bq,
                          const float* __restrict__ bk,
                          unsigned short* __restrict__ q2,
                          unsigned short* __restrict__ k2) {
    const int wave = threadIdx.x >> 5;
    const int lane = threadIdx.x & 31;
    const int half = lane >> 4;
    const int lr   = lane & 15;

    const bool isq = (blockIdx.y == 0);
    const int o0   = wave * 64;                    // this wave's 64 output cols
    const int row0 = blockIdx.x * 16;

    const unsigned short* Wb  = isq ? wqb : wkb;
    const float*          bia = isq ? bq : bk;

    v8f acc[4];
#pragma unroll
    for (int j = 0; j < 4; ++j)
#pragma unroll
        for (int r = 0; r < 8; ++r) acc[j][r] = 0.f;

#pragma unroll
    for (int kb = 0; kb < 16; ++kb) {
        const int k0 = kb * 32;
        v16bf a = loadA(sbf, HID, row0, k0, half, lr);
#pragma unroll
        for (int j = 0; j < 4; ++j) {
            v16bf b = loadBW(Wb, HID, o0 + j * 16, k0, half, lr);
            acc[j] = wmma_bf16(a, b, acc[j]);
        }
    }

    const float scale = isq ? 0.125f : 1.0f;       // DH^-0.5 = 0.125 for q
    unsigned short* dst = isq ? q2 : k2;
#pragma unroll
    for (int j = 0; j < 4; ++j) {
        const int   col = o0 + j * 16 + lr;
        const float bv  = bia[col];
        const int   hd  = col >> 6;
        const int   d   = col & 63;
#pragma unroll
        for (int r = 0; r < 8; ++r) {
            const int m = row0 + half * 8 + r;     // token row (i,t,b)
            const int b_ = m & 3, t = (m >> 2) & 127, i = m >> 9;
            const float v = (acc[j][r] + bv) * scale;
            dst[(((size_t)(i * BSZ + b_) * NH + hd) * NSEQ + t) * DH + d] = f2bf(v);
        }
    }
}

// ---------------------------------------------------------------------------
// K2: scores = q @ k^T per (i,b,head), mask, softmax over s, mean over heads
// -> w bf16 (i,b,t,s).  One block per (i,b,t-tile of 16).  64 KB LDS tile.
// ---------------------------------------------------------------------------
__global__ void scores_kernel(const unsigned short* __restrict__ q2,
                              const unsigned short* __restrict__ k2,
                              const unsigned char* __restrict__ mask,
                              unsigned short* __restrict__ wbf) {
    __shared__ float sc[NH][16][NSEQ];             // 64 KB

    const int wave = threadIdx.x >> 5;
    const int lane = threadIdx.x & 31;
    const int half = lane >> 4;
    const int lr   = lane & 15;

    const int ib    = blockIdx.x >> 3;
    const int ttile = blockIdx.x & 7;
    const int i = ib >> 2, b = ib & 3;
    const int t0 = ttile * 16;
    const int n0 = wave * 16;                      // this wave's s-tile

    const size_t base = (size_t)(i * BSZ + b) * NH * NSEQ * DH;
#pragma unroll
    for (int hd = 0; hd < NH; ++hd) {
        const unsigned short* qb = q2 + base + (size_t)hd * NSEQ * DH;
        const unsigned short* kb = k2 + base + (size_t)hd * NSEQ * DH;
        v8f acc = {0.f, 0.f, 0.f, 0.f, 0.f, 0.f, 0.f, 0.f};
#pragma unroll
        for (int kk = 0; kk < 2; ++kk) {           // K = 64 = 2 x 32
            v16bf a = loadA(qb, DH, t0, kk * 32, half, lr);
            v16bf bb = loadBW(kb, DH, n0, kk * 32, half, lr);
            acc = wmma_bf16(a, bb, acc);
        }
#pragma unroll
        for (int r = 0; r < 8; ++r)
            sc[hd][half * 8 + r][n0 + lr] = acc[r];
    }
    __syncthreads();

    // softmax over s per (head, t) row — 128 rows, one thread each
    const int tid = threadIdx.x;
    if (tid < 128) {
        const int hd = tid >> 4, tr = tid & 15;
        const unsigned char* mrow = mask + (size_t)(i * BSZ + b) * NSEQ;
        float mx = -__builtin_inff();
        for (int s = 0; s < NSEQ; ++s)
            if (!mrow[s]) mx = fmaxf(mx, sc[hd][tr][s]);
        float sum = 0.f;
        for (int s = 0; s < NSEQ; ++s) {
            float e = mrow[s] ? 0.f : __expf(sc[hd][tr][s] - mx);
            sc[hd][tr][s] = e;
            sum += e;
        }
        const float inv = 1.0f / sum;
        for (int s = 0; s < NSEQ; ++s) sc[hd][tr][s] *= inv;
    }
    __syncthreads();

    // mean over heads, emit bf16 w[i,b,t,s]
    for (int idx = tid; idx < 16 * NSEQ; idx += 256) {
        const int t = idx >> 7, s = idx & 127;
        float a = 0.f;
#pragma unroll
        for (int hd = 0; hd < NH; ++hd) a += sc[hd][t][s];
        wbf[((size_t)(i * BSZ + b) * NSEQ + t0 + t) * NSEQ + s] = f2bf(a * 0.125f);
    }
}

// ---------------------------------------------------------------------------
// Shared helper: stage a 32 x 512 bf16 slab (rows strided in global) into LDS.
// Uses CDNA5 async global->LDS DMA (ASYNCcnt) when the builtin is available.
// ---------------------------------------------------------------------------
static __device__ __forceinline__ void stage_slab(
        unsigned short* __restrict__ Bs,          // [32][HID]
        const unsigned short* __restrict__ gbase, // element ptr of (row 0, col 0)
        size_t row_stride_elems) {
    const int lin = threadIdx.x;
#pragma unroll
    for (int r = 0; r < 8; ++r) {
        const int chunk = lin + r * 256;          // 2048 chunks of 8 bf16
        const int srow = chunk >> 6;
        const int c8   = chunk & 63;
        const unsigned short* g = gbase + (size_t)srow * row_stride_elems + c8 * 8;
        unsigned short* l = &Bs[(size_t)srow * HID + c8 * 8];
#if HAVE_ASYNC_LDS
        // param0: AS(1) pointer to 16B int vector; param1: AS(3) ditto.
        __builtin_amdgcn_global_load_async_to_lds_b128(
            (__attribute__((address_space(1))) gv4i*)(unsigned long long)g,
            (__attribute__((address_space(3))) gv4i*)(unsigned)(unsigned long long)l,
            0, 0);
#else
        *reinterpret_cast<u32x4*>(l) = *reinterpret_cast<const u32x4*>(g);
#endif
    }
#if HAVE_ASYNC_LDS
    __builtin_amdgcn_s_wait_asynccnt(0);
#endif
}

// ---------------------------------------------------------------------------
// K3: x[i,t,b,h] = sum_s w[i,b,t,s] * state[i,s,b,h]
// One block per (i,b,t-tile). B operand staged through LDS (K-major reads).
// ---------------------------------------------------------------------------
__global__ void x_kernel(const unsigned short* __restrict__ wbf,
                         const unsigned short* __restrict__ sbf,
                         unsigned short* __restrict__ xbf) {
    __shared__ __align__(16) unsigned short Bs[32 * HID];   // 32 KB

    const int wave = threadIdx.x >> 5;
    const int lane = threadIdx.x & 31;
    const int half = lane >> 4;
    const int lr   = lane & 15;

    const int ib = blockIdx.x >> 3, ttile = blockIdx.x & 7;
    const int i = ib >> 2, b = ib & 3;
    const int t0 = ttile * 16;

    const unsigned short* wbase = wbf + (size_t)(i * BSZ + b) * NSEQ * NSEQ;

    v8f acc[4];
#pragma unroll
    for (int j = 0; j < 4; ++j)
#pragma unroll
        for (int r = 0; r < 8; ++r) acc[j][r] = 0.f;

    for (int kc = 0; kc < 4; ++kc) {              // K = 128 in 4 slabs of 32
        const int s0 = kc * 32;
        // state[i, s0+srow, b, :]  (row stride over s = BSZ*HID = 2048)
        stage_slab(Bs, sbf + ((size_t)(i * NSEQ + s0) * BSZ + b) * HID,
                   (size_t)BSZ * HID);
        __syncthreads();

        v16bf a = loadA(wbase, NSEQ, t0, s0, half, lr);
#pragma unroll
        for (int j = 0; j < 4; ++j) {
            const int n0 = wave * 64 + j * 16;
            v16bf bb = loadB_lds(Bs, n0 + lr, half);
            acc[j] = wmma_bf16(a, bb, acc[j]);
        }
        __syncthreads();
    }

#pragma unroll
    for (int j = 0; j < 4; ++j) {
        const int n0 = wave * 64 + j * 16;
#pragma unroll
        for (int r = 0; r < 8; ++r) {
            const int t = t0 + half * 8 + r;
            xbf[((size_t)(i * NSEQ + t) * BSZ + b) * HID + n0 + lr] = f2bf(acc[j][r]);
        }
    }
}

// ---------------------------------------------------------------------------
// K4: rel_mix[i,t,b,h] = sum_s w[i,b,t,s] * relation[s,t,b,h]
// One block per (t,b,i-tile). A rows = i (stride 65536 elems in wbf).
// ---------------------------------------------------------------------------
__global__ void rm_kernel(const unsigned short* __restrict__ wbf,
                          const unsigned short* __restrict__ rbf,
                          unsigned short* __restrict__ rmbf) {
    __shared__ __align__(16) unsigned short Bs[32 * HID];   // 32 KB

    const int wave = threadIdx.x >> 5;
    const int lane = threadIdx.x & 31;
    const int half = lane >> 4;
    const int lr   = lane & 15;

    const int tb = blockIdx.x >> 3, itile = blockIdx.x & 7;
    const int t = tb >> 2, b = tb & 3;
    const int i0 = itile * 16;

    // A[i][s] = wbf[((i*B+b)*N + t)*N + s]; row stride over i = B*N*N = 65536
    const unsigned short* abase =
        wbf + ((size_t)i0 * BSZ + b) * (NSEQ * NSEQ) + (size_t)t * NSEQ;

    v8f acc[4];
#pragma unroll
    for (int j = 0; j < 4; ++j)
#pragma unroll
        for (int r = 0; r < 8; ++r) acc[j][r] = 0.f;

    for (int kc = 0; kc < 4; ++kc) {
        const int s0 = kc * 32;
        // relation[s0+srow, t, b, :]  (row stride over s = N*B*H = 262144)
        stage_slab(Bs, rbf + ((size_t)(s0 * NSEQ + t) * BSZ + b) * HID,
                   (size_t)NSEQ * BSZ * HID);
        __syncthreads();

        v16bf a = loadA(abase, (long)BSZ * NSEQ * NSEQ, 0, s0, half, lr);
#pragma unroll
        for (int j = 0; j < 4; ++j) {
            const int n0 = wave * 64 + j * 16;
            v16bf bb = loadB_lds(Bs, n0 + lr, half);
            acc[j] = wmma_bf16(a, bb, acc[j]);
        }
        __syncthreads();
    }

#pragma unroll
    for (int j = 0; j < 4; ++j) {
        const int n0 = wave * 64 + j * 16;
#pragma unroll
        for (int r = 0; r < 8; ++r) {
            const int i = i0 + half * 8 + r;
            rmbf[((size_t)(i * NSEQ + t) * BSZ + b) * HID + n0 + lr] = f2bf(acc[j][r]);
        }
    }
}

// ---------------------------------------------------------------------------
// K5: out = relu([x, rel_mix] @ Wp^T + bp); y = state + out; LayerNorm(y)
// One block per 16-token tile; full 512-wide row kept in LDS for LN.
// ---------------------------------------------------------------------------
__global__ void final_kernel(const unsigned short* __restrict__ xbf,
                             const unsigned short* __restrict__ rmbf,
                             const unsigned short* __restrict__ wpb,
                             const float* __restrict__ bp,
                             const float* __restrict__ state,
                             const float* __restrict__ gamma,
                             const float* __restrict__ beta,
                             float* __restrict__ out) {
    __shared__ float yb[16][HID];                  // 32 KB
    __shared__ float smu[16], srs[16];

    const int wave = threadIdx.x >> 5;
    const int lane = threadIdx.x & 31;
    const int half = lane >> 4;
    const int lr   = lane & 15;
    const int m0   = blockIdx.x * 16;

    v8f acc[4];
#pragma unroll
    for (int j = 0; j < 4; ++j)
#pragma unroll
        for (int r = 0; r < 8; ++r) acc[j][r] = 0.f;

    // c in [0,512): x part against Wp[:, 0:512]
#pragma unroll
    for (int kb = 0; kb < 16; ++kb) {
        const int k0 = kb * 32;
        v16bf a = loadA(xbf, HID, m0, k0, half, lr);
#pragma unroll
        for (int j = 0; j < 4; ++j) {
            const int n0 = wave * 64 + j * 16;
            v16bf bb = loadBW(wpb, 2 * HID, n0, k0, half, lr);
            acc[j] = wmma_bf16(a, bb, acc[j]);
        }
    }
    // c in [512,1024): rel_mix part against Wp[:, 512:1024]
#pragma unroll
    for (int kb = 0; kb < 16; ++kb) {
        const int k0 = kb * 32;
        v16bf a = loadA(rmbf, HID, m0, k0, half, lr);
#pragma unroll
        for (int j = 0; j < 4; ++j) {
            const int n0 = wave * 64 + j * 16;
            v16bf bb = loadBW(wpb + HID, 2 * HID, n0, k0, half, lr);
            acc[j] = wmma_bf16(a, bb, acc[j]);
        }
    }

    // bias + relu + residual into LDS row tile
#pragma unroll
    for (int j = 0; j < 4; ++j) {
        const int col = wave * 64 + j * 16 + lr;
        const float bv = bp[col];
#pragma unroll
        for (int r = 0; r < 8; ++r) {
            const int row = half * 8 + r;
            float v = fmaxf(acc[j][r] + bv, 0.f);
            yb[row][col] = v + state[(size_t)(m0 + row) * HID + col];
        }
    }
    __syncthreads();

    // LayerNorm stats: one thread per row
    if (threadIdx.x < 16) {
        const int row = threadIdx.x;
        float s = 0.f, s2 = 0.f;
        for (int c = 0; c < HID; ++c) {
            const float v = yb[row][c];
            s += v; s2 += v * v;
        }
        const float mu = s * (1.0f / HID);
        const float var = s2 * (1.0f / HID) - mu * mu;
        smu[row] = mu;
        srs[row] = rsqrtf(var + LN_EPS);
    }
    __syncthreads();

    for (int idx = threadIdx.x; idx < 16 * HID; idx += 256) {
        const int row = idx >> 9, col = idx & 511;
        const float y = yb[row][col];
        out[(size_t)(m0 + row) * HID + col] =
            gamma[col] * (y - smu[row]) * srs[row] + beta[col];
    }
}

// ---------------------------------------------------------------------------
// Host-side launch
// ---------------------------------------------------------------------------
extern "C" void kernel_launch(void* const* d_in, const int* in_sizes, int n_in,
                              void* d_out, int out_size, void* d_ws, size_t ws_size,
                              hipStream_t stream) {
    (void)in_sizes; (void)n_in; (void)out_size; (void)ws_size;

    const float*         state    = (const float*)d_in[0];
    const float*         relation = (const float*)d_in[1];
    const unsigned char* mask     = (const unsigned char*)d_in[2];
    // d_in[3] = n (always 128)
    const float* Wq = (const float*)d_in[4];
    const float* bq = (const float*)d_in[5];
    const float* Wk = (const float*)d_in[6];
    const float* bk = (const float*)d_in[7];
    const float* Wp = (const float*)d_in[8];
    const float* bp = (const float*)d_in[9];
    const float* gamma = (const float*)d_in[10];
    const float* beta  = (const float*)d_in[11];
    float* out = (float*)d_out;

    // workspace carve-up (bytes)
    const size_t SB = (size_t)MTOK * HID;                // 33,554,432 elems
    char* ws = (char*)d_ws;
    unsigned short* sbf  = (unsigned short*)(ws);                    // state bf16
    unsigned short* rbf  = (unsigned short*)(ws + 2 * SB);           // relation bf16
    unsigned short* q2   = (unsigned short*)(ws + 4 * SB);           // q  (i,b,hd,t,d)
    unsigned short* k2   = (unsigned short*)(ws + 6 * SB);           // k  (i,b,hd,s,d)
    unsigned short* xbf  = (unsigned short*)(ws + 8 * SB);           // x  (i,t,b,h)
    unsigned short* rmbf = (unsigned short*)(ws + 10 * SB);          // rel_mix
    unsigned short* wbf  = (unsigned short*)(ws + 12 * SB);          // w (i,b,t,s)
    unsigned short* wqb  = (unsigned short*)(ws + 12 * SB + 16777216);
    unsigned short* wkb  = (unsigned short*)(ws + 12 * SB + 16777216 + 524288);
    unsigned short* wpb  = (unsigned short*)(ws + 12 * SB + 16777216 + 1048576);

    // K0: converts
    cvt_f32_bf16<<<(int)(SB / 4 / 256), 256, 0, stream>>>(state, sbf, (int)(SB / 4));
    cvt_f32_bf16<<<(int)(SB / 4 / 256), 256, 0, stream>>>(relation, rbf, (int)(SB / 4));
    cvt_f32_bf16<<<256, 256, 0, stream>>>(Wq, wqb, HID * HID / 4);
    cvt_f32_bf16<<<256, 256, 0, stream>>>(Wk, wkb, HID * HID / 4);
    cvt_f32_bf16<<<512, 256, 0, stream>>>(Wp, wpb, HID * 2 * HID / 4);

    // K1: q/k projection (scaled, biased, scattered per-head), 4 N-tiles/wave
    qk_kernel<<<dim3(MTOK / 16, 2), 256, 0, stream>>>(sbf, wqb, wkb, bq, bk, q2, k2);

    // K2: scores + softmax + head-mean -> w
    scores_kernel<<<NSEQ * BSZ * 8, 256, 0, stream>>>(q2, k2, mask, wbf);

    // K3/K4: w @ state, w @ relation
    x_kernel<<<NSEQ * BSZ * 8, 256, 0, stream>>>(wbf, sbf, xbf);
    rm_kernel<<<NSEQ * BSZ * 8, 256, 0, stream>>>(wbf, rbf, rmbf);

    // K5: output projection + relu + residual + layernorm
    final_kernel<<<MTOK / 16, 256, 0, stream>>>(xbf, rmbf, wpb, bp, state,
                                                gamma, beta, out);
}
